// HarmonicBlockWithLI_56289841381382
// MI455X (gfx1250) — matrix operand order repaired and checked by reading, not compiled, not run on hardware
//
#include <hip/hip_runtime.h>
#include <cstddef>

// Problem constants (match reference)
#define Bb   8
#define Tt   2048
#define Dd   1024
#define BTr  (Bb * Tt)          // 16384 rows of flattened activation matrix
#define BD   (Bb * Dd)          // 8192 recurrence channels
#define NCH  16                 // scan chunks
#define LCH  (Tt / NCH)         // 128 steps per chunk (2^7)
#define DT_  0.1f
#define LNE  1e-5f

typedef __bf16 bf;
typedef __attribute__((ext_vector_type(16))) __bf16 v16bf;
typedef __attribute__((ext_vector_type(8)))  __bf16 v8bf;
typedef __attribute__((ext_vector_type(8)))  float  v8f;

static __device__ inline v8f vzero8() {
  v8f z = {0.f,0.f,0.f,0.f,0.f,0.f,0.f,0.f};
  return z;
}

// ---- WMMA fragment loaders (CDNA5 wave32 layouts, CDNA5 ISA §7.12.2) ----
static __device__ inline v16bf load_a_frag(const bf* __restrict__ A, int lda,
                                           int m0, int k0, int lane) {
  const int half = lane >> 4;
  const int mr   = lane & 15;
  const bf* p = A + (size_t)(m0 + mr) * lda + (size_t)(k0 + half * 8);
  v8bf lo = *(const v8bf*)(p);
  v8bf hi = *(const v8bf*)(p + 16);
  return __builtin_shufflevector(lo, hi, 0,1,2,3,4,5,6,7,8,9,10,11,12,13,14,15);
}

static __device__ inline v16bf load_b_frag(const bf* __restrict__ W, int ldw,
                                           int n0, int k0, int lane) {
  const int half = lane >> 4;
  const int nc   = lane & 15;
  const bf* p = W + (size_t)(n0 + nc) * ldw + (size_t)(k0 + half * 8);
  v8bf lo = *(const v8bf*)(p);
  v8bf hi = *(const v8bf*)(p + 16);
  return __builtin_shufflevector(lo, hi, 0,1,2,3,4,5,6,7,8,9,10,11,12,13,14,15);
}

static __device__ inline v8f wmma_bf16(v16bf a, v16bf b, v8f c) {
  return __builtin_amdgcn_wmma_f32_16x16x32_bf16(false, a, false, b, (short)0, c,
                                                 false, false);
}

static __device__ inline void store_tile_f32(float* __restrict__ C, int ldc,
                                             int m0, int n0, int lane, v8f c) {
  const int half = lane >> 4;
  const int col  = n0 + (lane & 15);
#pragma unroll
  for (int r = 0; r < 8; ++r) {
    const int row = m0 + r + half * 8;
    C[(size_t)row * ldc + col] = c[r];
  }
}

// ---------------------------------------------------------------------------
__global__ __launch_bounds__(256) void cvt_f32_bf16(const float* __restrict__ src,
                                                    bf* __restrict__ dst, int n) {
  int i = blockIdx.x * 256 + threadIdx.x;
  if (i < n) dst[i] = (bf)src[i];
}

// ---------------------------------------------------------------------------
// Plain GEMM: C_f32[BTr,N] = A_bf16[BTr,K] @ W_bf16[N,K]^T
// block = 4 waves; wave -> 32 rows x 64 cols (8 accumulators, B frag feeds 2 WMMAs)
__global__ __launch_bounds__(128) void gemm_bf16_f32(const bf* __restrict__ A,
                                                     const bf* __restrict__ W,
                                                     float* __restrict__ C,
                                                     int K, int N) {
  const int lane = threadIdx.x & 31;
  const int wave = threadIdx.x >> 5;
  const int m0 = blockIdx.y * 128 + wave * 32;
  const int n0 = blockIdx.x * 64;
  v8f c0[4], c1[4];
#pragma unroll
  for (int i = 0; i < 4; ++i) { c0[i] = vzero8(); c1[i] = vzero8(); }
  for (int k = 0; k < K; k += 32) {
    v16bf a0 = load_a_frag(A, K, m0,      k, lane);
    v16bf a1 = load_a_frag(A, K, m0 + 16, k, lane);
#pragma unroll
    for (int i = 0; i < 4; ++i) {
      v16bf b = load_b_frag(W, K, n0 + 16 * i, k, lane);
      c0[i] = wmma_bf16(a0, b, c0[i]);
      c1[i] = wmma_bf16(a1, b, c1[i]);
    }
  }
#pragma unroll
  for (int i = 0; i < 4; ++i) {
    store_tile_f32(C, N, m0,      n0 + 16 * i, lane, c0[i]);
    store_tile_f32(C, N, m0 + 16, n0 + 16 * i, lane, c1[i]);
  }
}

// ---------------------------------------------------------------------------
// Dual GEMM + exact GELU, bf16 out: X = gelu(A1@W1^T + A2@W2^T)
__global__ __launch_bounds__(128) void gemm_dual_gelu_bf16(
    const bf* __restrict__ A1, const bf* __restrict__ W1,
    const bf* __restrict__ A2, const bf* __restrict__ W2,
    bf* __restrict__ Xout, int K, int N) {
  const int lane = threadIdx.x & 31;
  const int wave = threadIdx.x >> 5;
  const int m0 = blockIdx.y * 128 + wave * 32;
  const int n0 = blockIdx.x * 64;
  v8f c0[4], c1[4];
#pragma unroll
  for (int i = 0; i < 4; ++i) { c0[i] = vzero8(); c1[i] = vzero8(); }
  for (int k = 0; k < K; k += 32) {
    v16bf a0 = load_a_frag(A1, K, m0,      k, lane);
    v16bf a1 = load_a_frag(A1, K, m0 + 16, k, lane);
#pragma unroll
    for (int i = 0; i < 4; ++i) {
      v16bf b = load_b_frag(W1, K, n0 + 16 * i, k, lane);
      c0[i] = wmma_bf16(a0, b, c0[i]);
      c1[i] = wmma_bf16(a1, b, c1[i]);
    }
  }
  for (int k = 0; k < K; k += 32) {
    v16bf a0 = load_a_frag(A2, K, m0,      k, lane);
    v16bf a1 = load_a_frag(A2, K, m0 + 16, k, lane);
#pragma unroll
    for (int i = 0; i < 4; ++i) {
      v16bf b = load_b_frag(W2, K, n0 + 16 * i, k, lane);
      c0[i] = wmma_bf16(a0, b, c0[i]);
      c1[i] = wmma_bf16(a1, b, c1[i]);
    }
  }
  const int half = lane >> 4;
#pragma unroll
  for (int i = 0; i < 4; ++i) {
    const int col = n0 + 16 * i + (lane & 15);
#pragma unroll
    for (int r = 0; r < 8; ++r) {
      const int row0 = m0 + r + half * 8;
      float x0 = c0[i][r];
      float x1 = c1[i][r];
      float g0 = 0.5f * x0 * (1.0f + erff(x0 * 0.70710678118654752f));
      float g1 = 0.5f * x1 * (1.0f + erff(x1 * 0.70710678118654752f));
      Xout[(size_t)row0 * N + col]        = (bf)g0;
      Xout[(size_t)(row0 + 16) * N + col] = (bf)g1;
    }
  }
}

// ---------------------------------------------------------------------------
// GLU GEMM: pair a-tile (col n) with g-tile (col n+D) in-wave; emit pre-LN
// u_l = a*sigmoid(g) + u_prev, never materializing h [B,T,2D].
__global__ __launch_bounds__(128) void gemm_glu_residual(
    const bf* __restrict__ X, const bf* __restrict__ Wglu,
    const float* __restrict__ bias, const float* __restrict__ uprev,
    float* __restrict__ out, int K) {
  const int lane = threadIdx.x & 31;
  const int wave = threadIdx.x >> 5;
  const int m0 = blockIdx.y * 64 + wave * 16;
  const int n0 = blockIdx.x * 64;
  v8f ca[4], cg[4];
#pragma unroll
  for (int i = 0; i < 4; ++i) { ca[i] = vzero8(); cg[i] = vzero8(); }
  for (int k = 0; k < K; k += 32) {
    v16bf a = load_a_frag(X, K, m0, k, lane);
#pragma unroll
    for (int i = 0; i < 4; ++i) {
      ca[i] = wmma_bf16(a, load_b_frag(Wglu, K, n0 + 16 * i,      k, lane), ca[i]);
      cg[i] = wmma_bf16(a, load_b_frag(Wglu, K, n0 + 16 * i + Dd, k, lane), cg[i]);
    }
  }
  const int half = lane >> 4;
#pragma unroll
  for (int i = 0; i < 4; ++i) {
    const int col = n0 + 16 * i + (lane & 15);
    const float ba = bias[col];
    const float bg = bias[col + Dd];
#pragma unroll
    for (int r = 0; r < 8; ++r) {
      const int row = m0 + r + half * 8;
      float av = ca[i][r] + ba;
      float gv = cg[i][r] + bg;
      float glu = av * (1.0f / (1.0f + expf(-gv)));
      out[(size_t)row * Dd + col] = glu + uprev[(size_t)row * Dd + col];
    }
  }
}

// ---------------------------------------------------------------------------
// Chunked parallel scan for the harmonic oscillator.
// Step: (u,v) <- M (u,v) + (dt*bx, dt*dt*bx),  M = [[1,-dt*om],[dt,1-dt*dt*om]]
// (M is constant per channel -> exact 3-phase chunked scan, 16x parallelism.)
static __device__ inline float softplus_f(float x) {
  return (x > 20.f) ? x : log1pf(expf(x));
}

// Phase 1: per (chunk, channel) local terminal state from zero init.
__global__ __launch_bounds__(256) void osc_phase1(const float* __restrict__ Bx,
                                                  const float* __restrict__ omega_raw,
                                                  float* __restrict__ locals /*[NCH][BD][2]*/) {
  const int idx = blockIdx.x * 256 + threadIdx.x;    // [0, NCH*BD)
  const int c  = idx / BD;
  const int ch = idx % BD;
  const int b = ch >> 10, d = ch & (Dd - 1);
  const float om = softplus_f(omega_raw[d]);
  const float* src = Bx + (size_t)b * Tt * Dd + (size_t)(c * LCH) * Dd + d;
  float u = 0.f, v = 0.f;
  for (int t = 0; t < LCH; ++t) {
    __builtin_prefetch(src + (size_t)(t + 16) * Dd, 0, 0);
    float f = src[(size_t)t * Dd];
    u += DT_ * (f - om * v);
    v += DT_ * u;
  }
  locals[((size_t)c * BD + ch) * 2 + 0] = u;
  locals[((size_t)c * BD + ch) * 2 + 1] = v;
}

// Phase 2: stitch chunk boundaries: start[c+1] = M^LCH * start[c] + local[c].
__global__ __launch_bounds__(256) void osc_phase2(const float* __restrict__ omega_raw,
                                                  const float* __restrict__ locals,
                                                  float* __restrict__ starts /*[NCH][BD][2]*/) {
  const int ch = blockIdx.x * 256 + threadIdx.x;     // [0, BD)
  const int d = ch & (Dd - 1);
  const float om = softplus_f(omega_raw[d]);
  // M^(2^7) by repeated squaring (LCH = 128)
  float a = 1.f,  bm = -DT_ * om, cm = DT_, dm = 1.f - DT_ * DT_ * om;
#pragma unroll
  for (int s = 0; s < 7; ++s) {
    float na = a * a + bm * cm;
    float nb = a * bm + bm * dm;
    float nc = cm * a + dm * cm;
    float nd = cm * bm + dm * dm;
    a = na; bm = nb; cm = nc; dm = nd;
  }
  float u = 0.f, v = 0.f;
  for (int c = 0; c < NCH; ++c) {
    starts[((size_t)c * BD + ch) * 2 + 0] = u;
    starts[((size_t)c * BD + ch) * 2 + 1] = v;
    const float lu = locals[((size_t)c * BD + ch) * 2 + 0];
    const float lv = locals[((size_t)c * BD + ch) * 2 + 1];
    float nu = a * u + bm * v + lu;
    float nv = cm * u + dm * v + lv;
    u = nu; v = nv;
  }
}

// Phase 3: replay each chunk from its correct start state; write bf16 osc_y.
__global__ __launch_bounds__(256) void osc_phase3(const float* __restrict__ Bx,
                                                  const float* __restrict__ omega_raw,
                                                  const float* __restrict__ starts,
                                                  bf* __restrict__ osc_bf) {
  const int idx = blockIdx.x * 256 + threadIdx.x;
  const int c  = idx / BD;
  const int ch = idx % BD;
  const int b = ch >> 10, d = ch & (Dd - 1);
  const float om = softplus_f(omega_raw[d]);
  const size_t base = (size_t)b * Tt * Dd + (size_t)(c * LCH) * Dd + d;
  const float* src = Bx + base;
  bf* dst = osc_bf + base;
  float u = starts[((size_t)c * BD + ch) * 2 + 0];
  float v = starts[((size_t)c * BD + ch) * 2 + 1];
  for (int t = 0; t < LCH; ++t) {
    __builtin_prefetch(src + (size_t)(t + 16) * Dd, 0, 0);
    float f = src[(size_t)t * Dd];
    u += DT_ * (f - om * v);
    v += DT_ * u;
    dst[(size_t)t * Dd] = (bf)v;
  }
}

// ---------------------------------------------------------------------------
// Chunked parallel scan for the leaky integrator: u' = alpha*u + (1-alpha)*x.
__global__ __launch_bounds__(256) void li_phase1(const float* __restrict__ Xin,
                                                 const float* __restrict__ tau_mem,
                                                 float* __restrict__ locals /*[NCH][BD]*/) {
  const int idx = blockIdx.x * 256 + threadIdx.x;
  const int c  = idx / BD;
  const int ch = idx % BD;
  const int b = ch >> 10, d = ch & (Dd - 1);
  const float alpha = expf(-1.0f / fabsf(tau_mem[d]));
  const float beta  = 1.0f - alpha;
  const float* src = Xin + (size_t)b * Tt * Dd + (size_t)(c * LCH) * Dd + d;
  float u = 0.f;
  for (int t = 0; t < LCH; ++t) {
    __builtin_prefetch(src + (size_t)(t + 16) * Dd, 0, 0);
    u = u * alpha + src[(size_t)t * Dd] * beta;
  }
  locals[(size_t)c * BD + ch] = u;
}

__global__ __launch_bounds__(256) void li_phase2(const float* __restrict__ tau_mem,
                                                 const float* __restrict__ locals,
                                                 float* __restrict__ starts /*[NCH][BD]*/) {
  const int ch = blockIdx.x * 256 + threadIdx.x;
  const int d = ch & (Dd - 1);
  const float alpha = expf(-1.0f / fabsf(tau_mem[d]));
  float aL = alpha;
#pragma unroll
  for (int s = 0; s < 7; ++s) aL = aL * aL;          // alpha^128
  float u = 0.f;
  for (int c = 0; c < NCH; ++c) {
    starts[(size_t)c * BD + ch] = u;
    u = aL * u + locals[(size_t)c * BD + ch];
  }
}

__global__ __launch_bounds__(256) void li_phase3(const float* __restrict__ Xin,
                                                 const float* __restrict__ tau_mem,
                                                 const float* __restrict__ starts,
                                                 float* __restrict__ li_f32,
                                                 bf* __restrict__ li_bf) {
  const int idx = blockIdx.x * 256 + threadIdx.x;
  const int c  = idx / BD;
  const int ch = idx % BD;
  const int b = ch >> 10, d = ch & (Dd - 1);
  const float alpha = expf(-1.0f / fabsf(tau_mem[d]));
  const float beta  = 1.0f - alpha;
  const size_t base = (size_t)b * Tt * Dd + (size_t)(c * LCH) * Dd + d;
  const float* src = Xin + base;
  float* dstf = li_f32 + base;
  bf*    dstb = li_bf  + base;
  float u = starts[(size_t)c * BD + ch];
  for (int t = 0; t < LCH; ++t) {
    __builtin_prefetch(src + (size_t)(t + 16) * Dd, 0, 0);
    u = u * alpha + src[(size_t)t * Dd] * beta;
    dstf[(size_t)t * Dd] = u;
    dstb[(size_t)t * Dd] = (bf)u;
  }
}

// ---------------------------------------------------------------------------
// In-place LayerNorm over rows of D=1024 (block per row, 4 elems/thread)
__global__ __launch_bounds__(256) void layernorm_inplace(float* __restrict__ X,
                                                         const float* __restrict__ gamma,
                                                         const float* __restrict__ beta) {
  __shared__ float r1[256];
  __shared__ float r2[256];
  const int tid = threadIdx.x;
  float* x = X + (size_t)blockIdx.x * Dd;
  float v[4], s = 0.f, s2 = 0.f;
#pragma unroll
  for (int i = 0; i < 4; ++i) {
    v[i] = x[tid + 256 * i];
    s += v[i];
    s2 += v[i] * v[i];
  }
  r1[tid] = s; r2[tid] = s2;
  __syncthreads();
  for (int off = 128; off > 0; off >>= 1) {
    if (tid < off) { r1[tid] += r1[tid + off]; r2[tid] += r2[tid + off]; }
    __syncthreads();
  }
  const float mu  = r1[0] * (1.0f / Dd);
  const float var = r2[0] * (1.0f / Dd) - mu * mu;
  const float inv = rsqrtf(var + LNE);
#pragma unroll
  for (int i = 0; i < 4; ++i) {
    const int d = tid + 256 * i;
    x[d] = (v[i] - mu) * inv * gamma[d] + beta[d];
  }
}

// ---------------------------------------------------------------------------
extern "C" void kernel_launch(void* const* d_in, const int* in_sizes, int n_in,
                              void* d_out, int out_size, void* d_ws, size_t ws_size,
                              hipStream_t stream) {
  const float* u_prev    = (const float*)d_in[0];
  const float* B_w       = (const float*)d_in[1];
  const float* omega_raw = (const float*)d_in[2];
  const float* li_w      = (const float*)d_in[3];
  const float* tau_mem   = (const float*)d_in[4];
  const float* C_w       = (const float*)d_in[5];
  const float* D_w       = (const float*)d_in[6];
  const float* glu_w     = (const float*)d_in[7];
  const float* glu_b     = (const float*)d_in[8];
  const float* ln_gamma  = (const float*)d_in[9];
  const float* ln_beta   = (const float*)d_in[10];

  float* out_ul = (float*)d_out;                       // u_l  [B,T,D]
  const size_t BTD = (size_t)Bb * Tt * Dd;
  float* out_li = out_ul + BTD;                        // li_y [B,T,D]

  // Workspace layout (~175 MiB)
  char* wsp = (char*)d_ws;
  float* f32buf = (float*)wsp;         wsp += BTD * sizeof(float);           // Bx, then in_sum
  bf* bfX  = (bf*)wsp;                 wsp += BTD * sizeof(bf);              // osc_y, then x_l
  bf* bfU  = (bf*)wsp;                 wsp += BTD * sizeof(bf);              // u_prev bf16
  bf* bfL  = (bf*)wsp;                 wsp += BTD * sizeof(bf);              // li_y  bf16
  bf* wB   = (bf*)wsp;                 wsp += (size_t)Dd * Dd * sizeof(bf);
  bf* wLI  = (bf*)wsp;                 wsp += (size_t)Dd * Dd * sizeof(bf);
  bf* wC   = (bf*)wsp;                 wsp += (size_t)Dd * Dd * sizeof(bf);
  bf* wD   = (bf*)wsp;                 wsp += (size_t)Dd * Dd * sizeof(bf);
  bf* wGLU = (bf*)wsp;                 wsp += (size_t)2 * Dd * Dd * sizeof(bf);
  float* scanLoc   = (float*)wsp;      wsp += (size_t)2 * NCH * BD * sizeof(float);
  float* scanStart = (float*)wsp;      wsp += (size_t)2 * NCH * BD * sizeof(float);
  (void)ws_size; (void)n_in; (void)in_sizes; (void)out_size;

  // Weights (~12 MiB bf16) stay resident in the 192 MB L2 across all GEMMs.
  cvt_f32_bf16<<<(int)(BTD / 256), 256, 0, stream>>>(u_prev, bfU, (int)BTD);
  cvt_f32_bf16<<<Dd * Dd / 256, 256, 0, stream>>>(B_w,  wB,  Dd * Dd);
  cvt_f32_bf16<<<Dd * Dd / 256, 256, 0, stream>>>(li_w, wLI, Dd * Dd);
  cvt_f32_bf16<<<Dd * Dd / 256, 256, 0, stream>>>(C_w,  wC,  Dd * Dd);
  cvt_f32_bf16<<<Dd * Dd / 256, 256, 0, stream>>>(D_w,  wD,  Dd * Dd);
  cvt_f32_bf16<<<2 * Dd * Dd / 256, 256, 0, stream>>>(glu_w, wGLU, 2 * Dd * Dd);

  const dim3 gemm_grid(Dd / 64, BTr / 128);   // 32-row wave tiles
  const dim3 glu_grid (Dd / 64, BTr / 64);    // 16-row wave tiles (8 accs already)
  const dim3 gemm_block(128);
  const int scan_grid  = NCH * BD / 256;      // 512 blocks (4096 waves)
  const int chan_grid  = BD / 256;            // 32 blocks

  // 1) Bx = u_prev @ B_w^T
  gemm_bf16_f32<<<gemm_grid, gemm_block, 0, stream>>>(bfU, wB, f32buf, Dd, Dd);
  // 2) oscillator: 3-phase chunked linear scan -> osc_y (bf16)
  osc_phase1<<<scan_grid, 256, 0, stream>>>(f32buf, omega_raw, scanLoc);
  osc_phase2<<<chan_grid, 256, 0, stream>>>(omega_raw, scanLoc, scanStart);
  osc_phase3<<<scan_grid, 256, 0, stream>>>(f32buf, omega_raw, scanStart, bfX);
  // 3) in_sum = osc_y @ li_w^T
  gemm_bf16_f32<<<gemm_grid, gemm_block, 0, stream>>>(bfX, wLI, f32buf, Dd, Dd);
  // 4) leaky integrator: 3-phase chunked scan -> li_y (f32 out + bf16 feed)
  li_phase1<<<scan_grid, 256, 0, stream>>>(f32buf, tau_mem, scanLoc);
  li_phase2<<<chan_grid, 256, 0, stream>>>(tau_mem, scanLoc, scanStart);
  li_phase3<<<scan_grid, 256, 0, stream>>>(f32buf, tau_mem, scanStart, out_li, bfL);
  // 5) x_l = gelu(li_y@C_w^T + u_prev@D_w^T)  (bf16, reuses bfX)
  gemm_dual_gelu_bf16<<<gemm_grid, gemm_block, 0, stream>>>(bfL, wC, bfU, wD, bfX, Dd, Dd);
  // 6) fused GLU GEMM + residual -> pre-LN u_l in d_out
  gemm_glu_residual<<<glu_grid, gemm_block, 0, stream>>>(bfX, wGLU, glu_b, u_prev,
                                                         out_ul, Dd);
  // 7) in-place LayerNorm
  layernorm_inplace<<<BTr, 256, 0, stream>>>(out_ul, ln_gamma, ln_beta);
}
